// LatentRecurrentDepthModel_32323923869791
// MI455X (gfx1250) — compile-verified, matrix operand order checked by reading, not run on hardware
//
#include <hip/hip_runtime.h>
#include <stddef.h>
#include <stdint.h>

// ---------------- model dims ----------------
#define Bq   8
#define Sq   128
#define Dq   512
#define Vq   32768
#define Hq   8
#define Pq   2048
#define Lq   4
#define DKq  64
#define Mrows (Bq*Sq)          // 1024 token rows
#define STEPSq 3

#define KSTEP    32            // WMMA K per step
#define LDSTRIDE 40            // bf16 elems per LDS row: 32 + 8 pad (80B, 16B aligned, conflict-free)

// ---------------- vector types for WMMA ----------------
typedef __attribute__((ext_vector_type(16))) __bf16 v16bf;
typedef __attribute__((ext_vector_type(8)))  __bf16 v8bf;
typedef __attribute__((ext_vector_type(8)))  float  v8f;

// Load one lane's 16 bf16 elements of a 16x32 WMMA operand tile.
// Elements [0..7] at p, [8..15] at p+16 (CDNA5 16-bit operand layout).
__device__ __forceinline__ v16bf ld_tile16(const __bf16* __restrict__ p) {
  v8bf lo = *reinterpret_cast<const v8bf*>(p);
  v8bf hi = *reinterpret_cast<const v8bf*>(p + 16);
  return __builtin_shufflevector(lo, hi, 0,1,2,3,4,5,6,7,8,9,10,11,12,13,14,15);
}

// ---------------- CDNA5 async global->LDS copy (16B) ----------------
// VDST = VGPR holding LDS byte offset; VADDR = 64-bit global address.
// Tracked with ASYNCcnt (CDNA5 ISA).
__device__ __forceinline__ void async_cp16(uint32_t lds_off, const void* gptr) {
  asm volatile("global_load_async_to_lds_b128 %0, %1, off"
               :: "v"(lds_off), "v"((uint64_t)(uintptr_t)gptr)
               : "memory");
}
__device__ __forceinline__ void wait_async0() {
  asm volatile("s_wait_asynccnt 0x0" ::: "memory");
}

// =====================================================================
// WMMA GEMM:  C[M,N] (f32) = A[M,K](bf16) @ W[N,K](bf16)^T + bias[N]
// block = 256 threads = 8 waves; block tile = 128(M) x 64(N).
// A/B K-slices staged in double-buffered LDS via async copies; each wave
// computes one 16-row M tile x four 16-col N tiles per K step (4 wmma).
// Main loop manually unrolled 2x so both LDS buffer bases are constants.
// ACCUM: C += result; RELU: relu after bias; WBF: also write bf16 copy;
// SWIZ: remap output row r=(s*B+b) -> b*S+s (final logits transpose).
// =====================================================================
template<bool ACCUM, bool RELU, bool WBF, bool SWIZ>
__global__ __launch_bounds__(256)
void gemm_wmma_bf16(const __bf16* __restrict__ A, const __bf16* __restrict__ W,
                    const float* __restrict__ bias,
                    float* __restrict__ C, __bf16* __restrict__ Cbf,
                    int M, int N, int K) {
  // LDS: A0 | A1 (128 rows each) then B0 | B1 (64 rows each), row = LDSTRIDE bf16
  __shared__ __align__(16) __bf16 lds[(128 + 128 + 64 + 64) * LDSTRIDE];

  const int tid  = threadIdx.x;
  const int lane = tid & 31;
  const int wave = tid >> 5;
  const int mblk = blockIdx.x * 128;
  const int n0   = blockIdx.y * 64;
  const int lrow = lane & 15;
  const int koff = (lane >> 4) * 8;

  // low 32 bits of the generic pointer to a __shared__ object = LDS byte offset
  const uint32_t ldsbase = (uint32_t)(uintptr_t)(&lds[0]);

  // stage one 128x32 A slice + 64x32 B slice into buffer `buf` at k-origin `k`
  auto issue_copies = [&](int buf, int k) {
    const uint32_t aoff = ldsbase + (uint32_t)(buf * 128 * LDSTRIDE) * 2u;
    const uint32_t boff = ldsbase + (uint32_t)((256 + buf * 64) * LDSTRIDE) * 2u;
#pragma unroll
    for (int c = 0; c < 2; ++c) {              // A: 512 x 16B chunks / 256 thr
      const int chunk = tid * 2 + c;
      const int row   = chunk >> 2;            // 0..127
      const int kk    = (chunk & 3) * 8;       // 0,8,16,24
      async_cp16(aoff + (uint32_t)(row * LDSTRIDE + kk) * 2u,
                 A + (size_t)(mblk + row) * K + k + kk);
    }
    {                                           // B: 256 x 16B chunks / 256 thr
      const int row = tid >> 2;                 // 0..63
      const int kk  = (tid & 3) * 8;
      async_cp16(boff + (uint32_t)(row * LDSTRIDE + kk) * 2u,
                 W + (size_t)(n0 + row) * K + k + kk);
    }
  };

  v8f acc[4] = {};

  // one K-step: load ALL operand tiles first, then 4 back-to-back WMMAs
  // (lets the legalizer use partial s_wait_dscnt instead of 0 per WMMA)
  auto compute = [&](int buf) {
    const __bf16* As = lds + buf * (128 * LDSTRIDE)
                           + (wave * 16 + lrow) * LDSTRIDE + koff;
    const __bf16* Bs = lds + (256 + buf * 64) * LDSTRIDE
                           + lrow * LDSTRIDE + koff;
    v16bf a  = ld_tile16(As);
    v16bf b0 = ld_tile16(Bs);
    v16bf b1 = ld_tile16(Bs + 16 * LDSTRIDE);
    v16bf b2 = ld_tile16(Bs + 32 * LDSTRIDE);
    v16bf b3 = ld_tile16(Bs + 48 * LDSTRIDE);
    acc[0] = __builtin_amdgcn_wmma_f32_16x16x32_bf16(false, a, false, b0, (short)0, acc[0], false, false);
    acc[1] = __builtin_amdgcn_wmma_f32_16x16x32_bf16(false, a, false, b1, (short)0, acc[1], false, false);
    acc[2] = __builtin_amdgcn_wmma_f32_16x16x32_bf16(false, a, false, b2, (short)0, acc[2], false, false);
    acc[3] = __builtin_amdgcn_wmma_f32_16x16x32_bf16(false, a, false, b3, (short)0, acc[3], false, false);
  };

  const int KT = K / KSTEP;                    // even for all uses (16 or 64)

  issue_copies(0, 0);
  wait_async0();
  __syncthreads();

  for (int kt = 0; kt < KT; kt += 2) {
    // phase 0: compute buf0, prefill buf1
    if (kt + 1 < KT) issue_copies(1, (kt + 1) * KSTEP);
    compute(0);
    wait_async0();
    __syncthreads();
    // phase 1: compute buf1, prefill buf0
    if (kt + 1 < KT) {
      if (kt + 2 < KT) issue_copies(0, (kt + 2) * KSTEP);
      compute(1);
      wait_async0();
      __syncthreads();
    }
  }

  // C/D layout: lanes 0-15 -> N=lane, M=+0..7; lanes 16-31 -> N=lane-16, M=+8..15
  const int cn    = lane & 15;
  const int mbase = mblk + wave * 16 + (lane >> 4) * 8;
#pragma unroll
  for (int t = 0; t < 4; ++t) {
    const int col = n0 + t * 16 + cn;
    const float bv = bias ? bias[col] : 0.0f;
#pragma unroll
    for (int i = 0; i < 8; ++i) {
      const int row = mbase + i;
      const size_t orow = SWIZ ? (size_t)((row & (Bq - 1)) * Sq + (row >> 3))
                               : (size_t)row;
      const size_t idx = orow * (size_t)N + col;
      float v = acc[t][i] + bv;
      if (ACCUM) v += C[idx];
      if (RELU)  v = v > 0.0f ? v : 0.0f;
      C[idx] = v;
      if (WBF)   Cbf[idx] = (__bf16)v;
    }
  }
}

// =====================================================================
// f32 -> bf16 conversion (grid-stride)
// =====================================================================
__global__ __launch_bounds__(256)
void cvt_bf16(const float* __restrict__ src, __bf16* __restrict__ dst, size_t n) {
  for (size_t i = (size_t)blockIdx.x * 256 + threadIdx.x; i < n;
       i += (size_t)gridDim.x * 256)
    dst[i] = (__bf16)src[i];
}

// =====================================================================
// Embedding gather: x[s*B+b, :] = emb[tokens[b,s], :]  (f32 + bf16)
// =====================================================================
__global__ __launch_bounds__(128)
void embed_k(const int* __restrict__ tokens, const float* __restrict__ emb,
             float* __restrict__ xf, __bf16* __restrict__ xb) {
  const int bs = blockIdx.x;                 // b*S + s
  const int b = bs >> 7, s = bs & (Sq - 1);
  const int tok = tokens[bs];
  const int d = threadIdx.x * 4;
  const float4 e = *reinterpret_cast<const float4*>(emb + (size_t)tok * Dq + d);
  const size_t r = (size_t)(s * Bq + b) * Dq + d;
  xf[r + 0] = e.x; xf[r + 1] = e.y; xf[r + 2] = e.z; xf[r + 3] = e.w;
  xb[r + 0] = (__bf16)e.x; xb[r + 1] = (__bf16)e.y;
  xb[r + 2] = (__bf16)e.z; xb[r + 3] = (__bf16)e.w;
}

// =====================================================================
// Causal multihead attention, one workgroup per (b,h), thread = query s.
// Online (flash-style) softmax in f32. K/V staged in LDS (64 KB).
// =====================================================================
__global__ __launch_bounds__(128)
void attention_k(const float* __restrict__ qkv, float* __restrict__ O,
                 __bf16* __restrict__ Obf) {
  __shared__ float kls[Sq][DKq];
  __shared__ float vls[Sq][DKq];
  const int b = blockIdx.x >> 3;
  const int h = blockIdx.x & (Hq - 1);
  const int t = threadIdx.x;                 // query index s
  const size_t rowt = (size_t)(t * Bq + b) * (3 * Dq);
  const float* kp = qkv + rowt + Dq     + h * DKq;
  const float* vp = qkv + rowt + 2 * Dq + h * DKq;
#pragma unroll
  for (int d = 0; d < DKq; d += 4) {
    *reinterpret_cast<float4*>(&kls[t][d]) = *reinterpret_cast<const float4*>(kp + d);
    *reinterpret_cast<float4*>(&vls[t][d]) = *reinterpret_cast<const float4*>(vp + d);
  }
  __syncthreads();

  float q[DKq];
  const float* qp = qkv + rowt + h * DKq;
#pragma unroll
  for (int d = 0; d < DKq; d += 4)
    *reinterpret_cast<float4*>(&q[d]) = *reinterpret_cast<const float4*>(qp + d);

  float o[DKq];
#pragma unroll
  for (int d = 0; d < DKq; ++d) o[d] = 0.0f;

  float mx = -3.0e38f, sum = 0.0f;
  const float scale = 0.125f;                // 1/sqrt(64)
  for (int j = 0; j <= t; ++j) {             // causal
    float s = 0.0f;
#pragma unroll
    for (int d = 0; d < DKq; ++d) s += q[d] * kls[j][d];
    s *= scale;
    const float nm = fmaxf(mx, s);
    const float corr = __expf(mx - nm);
    const float e = __expf(s - nm);
    sum = sum * corr + e;
#pragma unroll
    for (int d = 0; d < DKq; ++d) o[d] = o[d] * corr + e * vls[j][d];
    mx = nm;
  }
  const float inv = 1.0f / sum;
  float* op = O + (size_t)(t * Bq + b) * Dq + h * DKq;
  __bf16* obp = Obf + (size_t)(t * Bq + b) * Dq + h * DKq;
#pragma unroll
  for (int d = 0; d < DKq; ++d) {
    const float v = o[d] * inv;
    op[d] = v;
    obp[d] = (__bf16)v;
  }
}

// =====================================================================
// Fused residual add + LayerNorm over D=512; block per row, 256 threads.
// =====================================================================
__global__ __launch_bounds__(256)
void add_layernorm_k(const float* __restrict__ A, const float* __restrict__ Bv,
                     const float* __restrict__ w, const float* __restrict__ bi,
                     float* __restrict__ Y, __bf16* __restrict__ Yb) {
  __shared__ float s1[256], s2[256];
  const size_t base = (size_t)blockIdx.x * Dq;
  const int t = threadIdx.x;
  const float v0 = A[base + t]       + Bv[base + t];
  const float v1 = A[base + t + 256] + Bv[base + t + 256];
  s1[t] = v0 + v1;
  s2[t] = v0 * v0 + v1 * v1;
  __syncthreads();
  for (int off = 128; off > 0; off >>= 1) {
    if (t < off) { s1[t] += s1[t + off]; s2[t] += s2[t + off]; }
    __syncthreads();
  }
  const float mean = s1[0] * (1.0f / Dq);
  const float var  = s2[0] * (1.0f / Dq) - mean * mean;
  const float r    = rsqrtf(var + 1e-5f);
  const float y0 = (v0 - mean) * r * w[t]       + bi[t];
  const float y1 = (v1 - mean) * r * w[t + 256] + bi[t + 256];
  Y[base + t]        = y0;  Yb[base + t]        = (__bf16)y0;
  Y[base + t + 256]  = y1;  Yb[base + t + 256]  = (__bf16)y1;
}

// =====================================================================
// LSTM pointwise cell: torch gate order i,f,g,o at offsets 0,D,2D,3D.
// =====================================================================
__device__ __forceinline__ float sigm(float x) { return 1.0f / (1.0f + __expf(-x)); }

__global__ __launch_bounds__(256)
void lstm_cell_k(const float* __restrict__ gates, float* __restrict__ c,
                 float* __restrict__ hf, __bf16* __restrict__ hb) {
  const size_t idx = (size_t)blockIdx.x * 256 + threadIdx.x;   // over M*D
  const size_t row = idx >> 9;                                  // /512
  const int    j   = (int)(idx & (Dq - 1));
  const float* g = gates + row * (4 * Dq);
  const float ig = sigm(g[j]);
  const float fg = sigm(g[Dq + j]);
  const float gg = tanhf(g[2 * Dq + j]);
  const float og = sigm(g[3 * Dq + j]);
  const float cn = fg * c[idx] + ig * gg;
  const float hn = og * tanhf(cn);
  c[idx]  = cn;
  hf[idx] = hn;
  hb[idx] = (__bf16)hn;
}

__global__ __launch_bounds__(256)
void lstm_init_k(float* __restrict__ c, __bf16* __restrict__ hb) {
  const size_t idx = (size_t)blockIdx.x * 256 + threadIdx.x;
  c[idx] = 0.0f;
  hb[idx] = (__bf16)0.0f;
}

__global__ __launch_bounds__(256)
void add_res_k(const float* __restrict__ a, const float* __restrict__ b,
               float* __restrict__ of, __bf16* __restrict__ ob) {
  const size_t idx = (size_t)blockIdx.x * 256 + threadIdx.x;
  const float v = a[idx] + b[idx];
  of[idx] = v;
  ob[idx] = (__bf16)v;
}

// =====================================================================
// host: orchestrate the whole forward pass
// =====================================================================
extern "C" void kernel_launch(void* const* d_in, const int* in_sizes, int n_in,
                              void* d_out, int out_size, void* d_ws, size_t ws_size,
                              hipStream_t stream) {
  (void)in_sizes; (void)n_in; (void)out_size; (void)ws_size;

  const int*   tokens     = (const int*)  d_in[0];
  const float* emb        = (const float*)d_in[1];
  const float* in_proj_w  = (const float*)d_in[2];
  const float* in_proj_b  = (const float*)d_in[3];
  const float* attn_out_w = (const float*)d_in[4];
  const float* attn_out_b = (const float*)d_in[5];
  const float* ln1_w      = (const float*)d_in[6];
  const float* ln1_b      = (const float*)d_in[7];
  const float* lstm_wih   = (const float*)d_in[8];
  const float* lstm_whh   = (const float*)d_in[9];
  const float* lstm_bih   = (const float*)d_in[10];
  const float* lstm_bhh   = (const float*)d_in[11];
  const float* ff_w1      = (const float*)d_in[12];
  const float* ff_b1      = (const float*)d_in[13];
  const float* ff_w2      = (const float*)d_in[14];
  const float* ff_b2      = (const float*)d_in[15];
  const float* ln2_w      = (const float*)d_in[16];
  const float* ln2_b      = (const float*)d_in[17];
  const float* head_w     = (const float*)d_in[18];
  const float* head_b     = (const float*)d_in[19];
  float* out = (float*)d_out;

  // ---- bump allocator over d_ws ----
  size_t off = 0;
  auto bump = [&](size_t bytes) -> void* {
    void* p = (void*)((char*)d_ws + off);
    off += (bytes + 255) & ~(size_t)255;
    return p;
  };
  const size_t N_INPROJ = (size_t)Lq * 3 * Dq * Dq;
  const size_t N_ATTNW  = (size_t)Lq * Dq * Dq;
  const size_t N_LSTMW  = (size_t)Lq * 4 * Dq * Dq;
  const size_t N_FFW    = (size_t)Lq * Pq * Dq;
  const size_t N_HEADW  = (size_t)Vq * Dq;
  const size_t N_ACT    = (size_t)Mrows * Dq;
  const size_t N_QKV    = (size_t)Mrows * 3 * Dq;
  const size_t N_GATE   = (size_t)Mrows * 4 * Dq;
  const size_t N_FFH    = (size_t)Mrows * Pq;

  __bf16* inproj_bf = (__bf16*)bump(N_INPROJ * 2);
  __bf16* attnw_bf  = (__bf16*)bump(N_ATTNW * 2);
  __bf16* wih_bf    = (__bf16*)bump(N_LSTMW * 2);
  __bf16* whh_bf    = (__bf16*)bump(N_LSTMW * 2);
  __bf16* w1_bf     = (__bf16*)bump(N_FFW * 2);
  __bf16* w2_bf     = (__bf16*)bump(N_FFW * 2);
  __bf16* head_bf   = (__bf16*)bump(N_HEADW * 2);

  float*  x_f   = (float*) bump(N_ACT * 4);
  __bf16* x_b   = (__bf16*)bump(N_ACT * 2);
  float*  qkv_f = (float*) bump(N_QKV * 4);
  float*  o_f   = (float*) bump(N_ACT * 4);
  __bf16* o_b   = (__bf16*)bump(N_ACT * 2);
  float*  proj_f= (float*) bump(N_ACT * 4);
  float*  gate_f= (float*) bump(N_GATE * 4);
  float*  c_f   = (float*) bump(N_ACT * 4);
  float*  h_f   = (float*) bump(N_ACT * 4);
  __bf16* h_b   = (__bf16*)bump(N_ACT * 2);
  float*  lat_f = (float*) bump(N_ACT * 4);
  __bf16* lat_b = (__bf16*)bump(N_ACT * 2);
  float*  ffh_f = (float*) bump(N_FFH * 4);
  __bf16* ffh_b = (__bf16*)bump(N_FFH * 2);
  float*  ffo_f = (float*) bump(N_ACT * 4);

  // ---- weights -> bf16 ----
  cvt_bf16<<<4096, 256, 0, stream>>>(in_proj_w,  inproj_bf, N_INPROJ);
  cvt_bf16<<<4096, 256, 0, stream>>>(attn_out_w, attnw_bf,  N_ATTNW);
  cvt_bf16<<<4096, 256, 0, stream>>>(lstm_wih,   wih_bf,    N_LSTMW);
  cvt_bf16<<<4096, 256, 0, stream>>>(lstm_whh,   whh_bf,    N_LSTMW);
  cvt_bf16<<<4096, 256, 0, stream>>>(ff_w1,      w1_bf,     N_FFW);
  cvt_bf16<<<4096, 256, 0, stream>>>(ff_w2,      w2_bf,     N_FFW);
  cvt_bf16<<<8192, 256, 0, stream>>>(head_w,     head_bf,   N_HEADW);

  // ---- embedding ----
  embed_k<<<Mrows, 128, 0, stream>>>(tokens, emb, x_f, x_b);

  const dim3 blk(256);
  for (int l = 0; l < Lq; ++l) {
    gemm_wmma_bf16<false,false,false,false><<<dim3(Mrows/128, (3*Dq)/64), blk, 0, stream>>>(
        x_b, inproj_bf + (size_t)l * 3 * Dq * Dq, in_proj_b + (size_t)l * 3 * Dq,
        qkv_f, nullptr, Mrows, 3 * Dq, Dq);

    attention_k<<<Bq * Hq, 128, 0, stream>>>(qkv_f, o_f, o_b);

    gemm_wmma_bf16<false,false,false,false><<<dim3(Mrows/128, Dq/64), blk, 0, stream>>>(
        o_b, attnw_bf + (size_t)l * Dq * Dq, attn_out_b + (size_t)l * Dq,
        proj_f, nullptr, Mrows, Dq, Dq);

    add_layernorm_k<<<Mrows, 256, 0, stream>>>(x_f, proj_f,
        ln1_w + (size_t)l * Dq, ln1_b + (size_t)l * Dq, x_f, x_b);

    // ---- latent LSTM recurrence (3 steps) ----
    lstm_init_k<<<N_ACT/256, 256, 0, stream>>>(c_f, h_b);
    const __bf16* lat_in = x_b;
    for (int s = 0; s < STEPSq; ++s) {
      gemm_wmma_bf16<false,false,false,false><<<dim3(Mrows/128, (4*Dq)/64), blk, 0, stream>>>(
          lat_in, wih_bf + (size_t)l * 4 * Dq * Dq, lstm_bih + (size_t)l * 4 * Dq,
          gate_f, nullptr, Mrows, 4 * Dq, Dq);
      gemm_wmma_bf16<true,false,false,false><<<dim3(Mrows/128, (4*Dq)/64), blk, 0, stream>>>(
          h_b, whh_bf + (size_t)l * 4 * Dq * Dq, lstm_bhh + (size_t)l * 4 * Dq,
          gate_f, nullptr, Mrows, 4 * Dq, Dq);
      lstm_cell_k<<<N_ACT/256, 256, 0, stream>>>(gate_f, c_f, h_f, h_b);
      lat_in = h_b;
    }
    add_res_k<<<N_ACT/256, 256, 0, stream>>>(x_f, h_f, lat_f, lat_b);

    // ---- feed forward ----
    gemm_wmma_bf16<false,true,true,false><<<dim3(Mrows/128, Pq/64), blk, 0, stream>>>(
        lat_b, w1_bf + (size_t)l * Pq * Dq, ff_b1 + (size_t)l * Pq,
        ffh_f, ffh_b, Mrows, Pq, Dq);
    gemm_wmma_bf16<false,false,false,false><<<dim3(Mrows/128, Dq/64), blk, 0, stream>>>(
        ffh_b, w2_bf + (size_t)l * Dq * Pq, ff_b2 + (size_t)l * Dq,
        ffo_f, nullptr, Mrows, Dq, Pq);

    add_layernorm_k<<<Mrows, 256, 0, stream>>>(lat_f, ffo_f,
        ln2_w + (size_t)l * Dq, ln2_b + (size_t)l * Dq, x_f, x_b);
  }

  // ---- head: logits (B,S,V) with (s*B+b) -> (b*S+s) row swizzle ----
  gemm_wmma_bf16<false,false,false,true><<<dim3(Mrows/128, Vq/64), blk, 0, stream>>>(
      x_b, head_bf, head_b, out, nullptr, Mrows, Vq, Dq);
}
